// MemoryGraph_15453292331249
// MI455X (gfx1250) — matrix operand order repaired and checked by reading, not compiled, not run on hardware
//
#include <hip/hip_runtime.h>
#include <math.h>

// ---------------------------------------------------------------------------
// MI455X / gfx1250, wave32. Matrix work via v_wmma_f32_16x16x32_bf16.
// fc1_w (335 MB) streaming is the roofline term; W tiles are staged in LDS
// pre-packed in the WMMA-B bf16 register layout so fragment reads are pure
// ds_load_b32 with no conversion VALU in the hot loop.  A-fragment reads are
// unconditional (clamped address + cndmask zeroing) to avoid EXEC-masked
// per-element loads.
// ---------------------------------------------------------------------------

typedef __attribute__((ext_vector_type(16))) __bf16 v16bf;
typedef __attribute__((ext_vector_type(8)))  float  v8f;

#define Nn     8192
#define Kn     32
#define Dn     32
#define Tn     16
#define Cn     64
#define Hn     64
#define MODIN  160
#define STRIDE_ 4
#define WB_STRIDE 66                       // u32 row stride: conflict-free B reads
#define WB_WORDS (16 * WB_STRIDE)          // 16 K-pair rows x 66

__device__ __forceinline__ unsigned short f2bf(float f) {
  unsigned int x = __float_as_uint(f);
  unsigned int r = x + 0x7fffu + ((x >> 16) & 1u);   // round-to-nearest-even
  return (unsigned short)(r >> 16);
}
__device__ __forceinline__ float wsum32(float v) {
  #pragma unroll
  for (int m = 16; m >= 1; m >>= 1) v += __shfl_xor(v, m, 32);
  return v;
}
__device__ __forceinline__ float sigm(float x) { return 1.0f / (1.0f + __expf(-x)); }

// ---------------------------------------------------------------------------
// Kernel A: per-neuron modulator.  fc1 (160x64 per neuron) via WMMA bf16,
// then fc2 -> gates -> eff_prim / eff_decay / routing.  One wave per neuron;
// all LDS regions are wave-private (no block barriers needed).
// ---------------------------------------------------------------------------
__global__ __launch_bounds__(128) void modulator_kernel(
    const float* __restrict__ h_prev,   const float* __restrict__ prev_msg,
    const float* __restrict__ trace_prim, const float* __restrict__ trace_key,
    const float* __restrict__ primitives, const float* __restrict__ key_w,
    const float* __restrict__ decay_logit,
    const float* __restrict__ fc1_w,    const float* __restrict__ fc1_b,
    const float* __restrict__ fc2_w,    const float* __restrict__ fc2_b,
    const float* __restrict__ mod_lr_logit, const int* __restrict__ conn,
    float* __restrict__ effp, float* __restrict__ routing, float* __restrict__ effd)
{
  __shared__ unsigned int sWB[4 * WB_WORDS];     // packed-bf16 W tiles
  __shared__ float        sMI[4 * 2 * MODIN];    // mod_in rows
  __shared__ float        sXL[4 * 2 * Hn];       // x = tanh(fc1)

  const int lane = threadIdx.x & 31;
  const int wv   = threadIdx.x >> 5;
  const int n    = blockIdx.x * 4 + wv;

  unsigned int* wtB = &sWB[wv * WB_WORDS];
  float*        mi  = &sMI[wv * 2 * MODIN];
  float*        xl  = &sXL[wv * 2 * Hn];

  // per-lane (lane == d) vectors kept for dir()/eff computations
  float tp0 = trace_prim[(0 * Nn + n) * Dn + lane];
  float tp1 = trace_prim[(1 * Nn + n) * Dn + lane];
  float tk0 = trace_key [(0 * Nn + n) * Dn + lane];
  float tk1 = trace_key [(1 * Nn + n) * Dn + lane];
  float pr  = primitives[n * Dn + lane];
  float kw  = key_w     [n * Dn + lane];

  // mod_in = [h_prev, trace_prim, trace_key, primitives, key_w] into LDS
  #pragma unroll
  for (int b = 0; b < 2; b++) {
    mi[b * MODIN +   0 + lane] = h_prev[(b * Nn + n) * Dn + lane];
    mi[b * MODIN +  32 + lane] = b ? tp1 : tp0;
    mi[b * MODIN +  64 + lane] = b ? tk1 : tk0;
    mi[b * MODIN +  96 + lane] = pr;
    mi[b * MODIN + 128 + lane] = kw;
  }

  v8f acc[4] = {};                        // 4 h-tiles of the 16x64 f32 accumulator

  const int row    = lane & 15;           // A: M index
  const int rsel   = row & 1;             // clamped, always-valid row (0/1)
  const bool rlive = (row < 2);           // rows 2..15 of A are zero
  const int kbaseA = (lane >> 4) ? 16 : 0; // A swizzle half
  const int jbaseB = (lane >> 4) ? 8 : 0; // B: K-pair row base (khalf/2)
  const int col    = lane & 15;           // B: N index

  for (int kt = 0; kt < 5; kt++) {
    // ---- A fragment: unconditional LDS reads + cndmask zeroing ----
    union { v16bf v; unsigned short u[16]; } af;
    #pragma unroll
    for (int e = 0; e < 16; e++) {
      int kk = kt * 32 + kbaseA + e;
      float a = mi[rsel * MODIN + kk];    // always-valid address, no EXEC mask
      a = rlive ? a : 0.0f;               // v_cndmask
      af.u[e] = f2bf(a);
    }

    // ---- stage 32x64 fc1_w tile into LDS, converted + packed into the
    //      WMMA-B layout: dword wtB[j][h] = {lo: K=2j, hi: K=2j+1} ----
    const float* gw = fc1_w + ((long)n * MODIN + kt * 32) * Hn;
    unsigned short* wtB16 = (unsigned short*)wtB;
    #pragma unroll
    for (int i = 0; i < 16; i++) {
      int f4 = i * 32 + lane;             // float4 index 0..511
      int r  = f4 >> 4;                   // tile row (= K within tile) 0..31
      int c4 = (f4 & 15) * 4;             // tile col 0..60
      float4 v = ((const float4*)gw)[f4];
      int jp   = r >> 1;                  // K-pair row
      int half = r & 1;                   // low/high 16 bits of the dword
      unsigned short* p = wtB16 + ((jp * WB_STRIDE + c4) * 2 + half);
      p[0] = f2bf(v.x); p[2] = f2bf(v.y); p[4] = f2bf(v.z); p[6] = f2bf(v.w);
    }
    if (kt < 4) __builtin_prefetch(gw + 32 * Hn, 0, 1);   // global_prefetch_b8

    // ---- 4 h-tiles: B fragment = 8 direct b32 LDS reads (no VALU) + WMMA ----
    #pragma unroll
    for (int ht = 0; ht < 4; ht++) {
      union { v16bf v; unsigned int u[8]; } bfr;
      #pragma unroll
      for (int j = 0; j < 8; j++)
        bfr.u[j] = wtB[(jbaseB + j) * WB_STRIDE + ht * 16 + col];
      acc[ht] = __builtin_amdgcn_wmma_f32_16x16x32_bf16(
          false, af.v, false, bfr.v, (short)0, acc[ht], false, false);
    }
  }

  // ---- x = tanh(fc1 + b): rows M=0,1 live in VGPR0/1 of lanes 0..15 ----
  if (lane < 16) {
    #pragma unroll
    for (int ht = 0; ht < 4; ht++) {
      int h = ht * 16 + lane;
      float bias = fc1_b[n * Hn + h];
      xl[0 * Hn + h] = tanhf(acc[ht][0] + bias);
      xl[1 * Hn + h] = tanhf(acc[ht][1] + bias);
    }
  }

  // ---- fc2: 6 outputs (b,o); lanes 0..5 each do a 64-dot ----
  float o3v = 0.0f;
  if (lane < 6) {
    int b = lane / 3, o = lane - 3 * b;
    float s = fc2_b[n * 3 + o];
    for (int h = 0; h < Hn; h++)
      s += xl[b * Hn + h] * fc2_w[((long)n * Hn + h) * 3 + o];
    o3v = s;
  }
  float gp0 = tanhf(__shfl(o3v, 0, 32));
  float gk0 = tanhf(__shfl(o3v, 1, 32));
  float dm0 =        __shfl(o3v, 2, 32);
  float gp1 = tanhf(__shfl(o3v, 3, 32));
  float gk1 = tanhf(__shfl(o3v, 4, 32));
  float dm1 =        __shfl(o3v, 5, 32);

  float mlr = sigm(mod_lr_logit[0]);

  // dir() norms via wave32 reductions (clip at 1e-8)
  float np0 = fmaxf(sqrtf(wsum32(tp0 * tp0)), 1e-8f);
  float np1 = fmaxf(sqrtf(wsum32(tp1 * tp1)), 1e-8f);
  float nk0 = fmaxf(sqrtf(wsum32(tk0 * tk0)), 1e-8f);
  float nk1 = fmaxf(sqrtf(wsum32(tk1 * tk1)), 1e-8f);

  float ep0 = pr + mlr * gp0 * (tp0 / np0);
  float ep1 = pr + mlr * gp1 * (tp1 / np1);
  float ek0 = kw + mlr * gk0 * (tk0 / nk0);
  float ek1 = kw + mlr * gk1 * (tk1 / nk1);
  effp[(0 * Nn + n) * Dn + lane] = ep0;
  effp[(1 * Nn + n) * Dn + lane] = ep1;

  float dl = decay_logit[n];
  if (lane == 0) {
    effd[0 * Nn + n] = sigm(dl + dm0);
    effd[1 * Nn + n] = sigm(dl + dm1);
  }

  // routing[b,n,k] = sigmoid(eff_key . prev_msg[conn[n,k]])
  float r0 = 0.f, r1 = 0.f;
  for (int k = 0; k < Kn; k++) {
    int idx = conn[n * Kn + k];
    float s0 = wsum32(ek0 * prev_msg[(0 * Nn + idx) * Dn + lane]);
    float s1 = wsum32(ek1 * prev_msg[(1 * Nn + idx) * Dn + lane]);
    if (lane == k) { r0 = sigm(s0); r1 = sigm(s1); }
  }
  routing[(0 * Nn + n) * Kn + lane] = r0;
  routing[(1 * Nn + n) * Kn + lane] = r1;
}

// ---------------------------------------------------------------------------
// Step kernel: one scan step (launched 4x for grid-wide sync).  One wave per
// neuron, lane == d.  Dendrite tree is elementwise in d -> pure per-lane FMA;
// branch_w streamed once, reused for both batches.
// ---------------------------------------------------------------------------
__global__ __launch_bounds__(256) void step_kernel(
    const float* __restrict__ cc,   const int* __restrict__ conn,
    const float* __restrict__ bw,   const float* __restrict__ gw,
    const float* __restrict__ effp, const float* __restrict__ effd,
    const float* __restrict__ routing,
    const float* __restrict__ msrc, float* __restrict__ mdst,
    const float* __restrict__ hsrc, float* __restrict__ hdst,
    float* __restrict__ out, int t)
{
  const int lane = threadIdx.x & 31;
  const int n    = blockIdx.x * 8 + (threadIdx.x >> 5);

  float r0 = routing[(0 * Nn + n) * Kn + lane];
  float r1 = routing[(1 * Nn + n) * Kn + lane];

  float a0[4] = {0, 0, 0, 0}, a1[4] = {0, 0, 0, 0};
  #pragma unroll
  for (int k = 0; k < Kn; k++) {
    int idx  = conn[n * Kn + k];
    float w  = bw[(((long)n * 4 + (k >> 3)) * 8 + (k & 7)) * Dn + lane];
    float m0 = msrc[(0 * Nn + idx) * Dn + lane];
    float m1 = msrc[(1 * Nn + idx) * Dn + lane];
    a0[k >> 3] += __shfl(r0, k, 32) * m0 * w;
    a1[k >> 3] += __shfl(r1, k, 32) * m1 * w;
  }
  float g0 = 0.f, g1 = 0.f;
  #pragma unroll
  for (int v = 0; v < 4; v++) {
    float gv = gw[((long)n * 4 + v) * Dn + lane];
    g0 += tanhf(a0[v]) * gv;
    g1 += tanhf(a1[v]) * gv;
  }
  float rec0 = tanhf(g0), rec1 = tanhf(g1);   // NG==1 -> mean is identity
  if (n < Cn) {
    rec0 += cc[((0 * Tn + t * STRIDE_) * Cn + n) * Dn + lane];
    rec1 += cc[((1 * Tn + t * STRIDE_) * Cn + n) * Dn + lane];
  }
  float ed0 = effd[0 * Nn + n], ed1 = effd[Nn + n];
  float h0 = hsrc[(0 * Nn + n) * Dn + lane];
  float h1 = hsrc[(1 * Nn + n) * Dn + lane];
  h0 = ed0 * h0 + (1.f - ed0) * rec0;
  h1 = ed1 * h1 + (1.f - ed1) * rec1;
  hdst[(0 * Nn + n) * Dn + lane] = h0;
  hdst[(1 * Nn + n) * Dn + lane] = h1;
  float ms0 = tanhf(h0 * effp[(0 * Nn + n) * Dn + lane]);
  float ms1 = tanhf(h1 * effp[(1 * Nn + n) * Dn + lane]);
  mdst[(0 * Nn + n) * Dn + lane] = ms0;
  mdst[(1 * Nn + n) * Dn + lane] = ms1;
  if (n < Cn) {
    #pragma unroll
    for (int r = 0; r < STRIDE_; r++) {     // output replicated stride x
      out[((0 * Tn + t * STRIDE_ + r) * Cn + n) * Dn + lane] = ms0;
      out[((1 * Tn + t * STRIDE_ + r) * Cn + n) * Dn + lane] = ms1;
    }
  }
}

// ---------------------------------------------------------------------------
extern "C" void kernel_launch(void* const* d_in, const int* in_sizes, int n_in,
                              void* d_out, int out_size, void* d_ws, size_t ws_size,
                              hipStream_t stream)
{
  const float* cc         = (const float*)d_in[0];
  const float* h_prev     = (const float*)d_in[1];
  const float* prev_msg   = (const float*)d_in[2];
  const float* trace_prim = (const float*)d_in[3];
  const float* trace_key  = (const float*)d_in[4];
  const float* primitives = (const float*)d_in[5];
  const float* key_w      = (const float*)d_in[6];
  const float* decay_log  = (const float*)d_in[7];
  const float* bw         = (const float*)d_in[8];
  const float* gw         = (const float*)d_in[9];
  const float* fc1w       = (const float*)d_in[10];
  const float* fc1b       = (const float*)d_in[11];
  const float* fc2w       = (const float*)d_in[12];
  const float* fc2b       = (const float*)d_in[13];
  const float* mlr        = (const float*)d_in[14];
  const int*   conn       = (const int*)d_in[15];
  (void)in_sizes; (void)n_in; (void)out_size; (void)ws_size;

  float* ws      = (float*)d_ws;
  float* effp    = ws;                       // 2*N*D
  float* routing = effp + 2 * Nn * Dn;       // 2*N*K
  float* effd    = routing + 2 * Nn * Kn;    // 2*N
  float* hbuf    = effd + 2 * Nn;            // 2*N*D
  float* msgA    = hbuf + 2 * Nn * Dn;       // 2*N*D
  float* msgB    = msgA + 2 * Nn * Dn;       // 2*N*D
  float* out     = (float*)d_out;

  modulator_kernel<<<Nn / 4, 128, 0, stream>>>(
      h_prev, prev_msg, trace_prim, trace_key, primitives, key_w, decay_log,
      fc1w, fc1b, fc2w, fc2b, mlr, conn, effp, routing, effd);

  const float* msrc = prev_msg;  float* mdst = msgA;
  const float* hs   = h_prev;    float* hd   = hbuf;
  for (int t = 0; t < 4; t++) {
    step_kernel<<<Nn / 8, 256, 0, stream>>>(
        cc, conn, bw, gw, effp, effd, routing, msrc, mdst, hs, hd, out, t);
    msrc = mdst;
    mdst = (mdst == msgA) ? msgB : msgA;
    hs = hbuf; hd = hbuf;                    // in-place h update after step 0
  }
}
